// STU_8048768713011
// MI455X (gfx1250) — matrix operand order repaired and checked by reading, compiled, not run on hardware
//
#include <hip/hip_runtime.h>

// Problem constants (match reference)
#define SEQ_LEN 4096
#define DIM     1024
#define KF      24
#define BATCH   4

typedef __attribute__((ext_vector_type(16))) __bf16 v16bf;
typedef __attribute__((ext_vector_type(8)))  __bf16 v8bf;
typedef __attribute__((ext_vector_type(8)))  float  v8f;
typedef __attribute__((ext_vector_type(4)))  float  v4f;

// Workspace layout (bytes):
//  [0, XPROJ_BYTES)           : x_proj TRANSPOSED, bf16, shape (B, D, L)
//  [AFRAG_OFF, +AFRAG_BYTES)  : conv A-fragment table
//        Afrag[diff_idx][k][lane][e], diff = 16*diff_idx (0..255)
//  [MIFRAG_OFF, +MIFRAG_BYTES): M_inputs pre-packed as WMMA B fragments
static constexpr size_t XPROJ_BYTES  = (size_t)BATCH * SEQ_LEN * DIM * 2;
static constexpr size_t AFRAG_OFF    = XPROJ_BYTES;
static constexpr size_t AFRAG_BYTES  = (size_t)256 * KF * 32 * 16 * 2;
static constexpr size_t MIFRAG_OFF   = AFRAG_OFF + AFRAG_BYTES;
static constexpr size_t MIFRAG_BYTES = (size_t)64 * 32 * 32 * 16 * 2;

static constexpr int ABLK_BYTES = KF * 512 * 2;   // 24 KB per A block

// ---------------------------------------------------------------------------
// Stage one 24 KB A-fragment block global->LDS.
// Preferred path: Tensor Data Mover (one D#-driven DMA issued by wave 0,
// tracked by TENSORcnt). Fallback: per-lane async b128 copies (ASYNCcnt).
// ---------------------------------------------------------------------------
#if __has_builtin(__builtin_amdgcn_tensor_load_to_lds)
#define USE_TDM 1
typedef unsigned int u32x4 __attribute__((ext_vector_type(4)));
typedef int          i32x8 __attribute__((ext_vector_type(8)));
typedef int          i32x4 __attribute__((ext_vector_type(4)));
#else
#define USE_TDM 0
#endif

__device__ __forceinline__ void stage_ablk(const __bf16* __restrict__ src,
                                           __bf16* lds_dst, int tid, int wave) {
#if USE_TDM
  if (wave == 0) {
    const unsigned long long g = (unsigned long long)src;
    const unsigned lds = (unsigned)(unsigned long long)lds_dst;  // LDS byte addr
    // D# group 0: count=1 | lds_addr | global_addr | type=2 (bits 127:126)
    u32x4 g0;
    g0[0] = 1u;
    g0[1] = lds;
    g0[2] = (unsigned)(g & 0xffffffffu);
    g0[3] = (unsigned)((g >> 32) & 0x01ffffffu) | 0x80000000u;
    // D# group 1: data_size=8B (code 3); 1-D tile of 3072 elems;
    // tensor_dim0 = stride = 3072; everything else 0.
    enum { ELEMS = ABLK_BYTES / 8 };                 // 3072
    i32x8 g1;
    g1[0] = (3 << 16);                               // data_size = 8 bytes
    g1[1] = (int)((unsigned)ELEMS << 16);            // tensor_dim0[15:0]
    g1[2] = (int)(1u << 16);                         // dim0[31:16]=0, tensor_dim1=1
    g1[3] = (int)((unsigned)ELEMS << 16);            // dim1[31:16]=0, tile_dim0
    g1[4] = 0;                                       // tile_dim1=0, tile_dim2=0
    g1[5] = ELEMS;                                   // tensor_dim0_stride[31:0]
    g1[6] = 0;
    g1[7] = 0;
    i32x4 g2 = {0, 0, 0, 0};                         // VADDR2 group (unused, 1-D)
    i32x4 g3 = {0, 0, 0, 0};                         // VADDR3 group (unused, 1-D)
    i32x8 g4 = {0, 0, 0, 0, 0, 0, 0, 0};             // extra group (6-arg form)
    __builtin_amdgcn_tensor_load_to_lds(g0, g1, g2, g3, g4, 0);
  }
#else
#pragma unroll
  for (int i = 0; i < 6; ++i) {
    const unsigned off = (unsigned)(tid + i * 256) * 16u;
    unsigned long long g = (unsigned long long)(const char*)src + off;
    unsigned l = (unsigned)(unsigned long long)lds_dst + off;
    asm volatile("global_load_async_to_lds_b128 %0, %1, off"
                 :: "v"(l), "v"(g) : "memory");
  }
#endif
}

__device__ __forceinline__ void stage_wait() {
#if USE_TDM
  __builtin_amdgcn_s_wait_tensorcnt(0);   // nop for waves with TENSORcnt==0
#else
  asm volatile("s_wait_asynccnt 0" ::: "memory");
#endif
  __syncthreads();
}

// ---------------------------------------------------------------------------
// Kernel 1: pack M_inputs (fp32) into bf16 WMMA B-fragment images.
// ---------------------------------------------------------------------------
__global__ __launch_bounds__(256) void stu_mifrag_build(
    const float* __restrict__ Mi, unsigned short* __restrict__ mifrag_raw) {
  __bf16* __restrict__ mifrag = (__bf16*)mifrag_raw;
  const int idx = blockIdx.x * blockDim.x + threadIdx.x;  // (tn, jb, lane)
  if (idx >= 64 * 32 * 32) return;
  const int lane = idx & 31;
  const int jb   = (idx >> 5) & 31;
  const int tn   = idx >> 10;
  const int h = lane >> 4, m = lane & 15;
  __bf16* dst = mifrag + (size_t)idx * 16;
#pragma unroll
  for (int e = 0; e < 16; ++e)
    dst[e] = (__bf16)Mi[(size_t)(jb * 32 + (h ? 8 : 0) + (e < 8 ? e : e + 8)) * DIM + tn * 16 + m];
}

// ---------------------------------------------------------------------------
// Kernel 2: conv A-fragment table (causality + even-lag mask + 2x baked in).
// ---------------------------------------------------------------------------
__global__ __launch_bounds__(256) void stu_afrag_build(
    const float* __restrict__ stu, unsigned short* __restrict__ afrag_raw) {
  __bf16* __restrict__ afrag = (__bf16*)afrag_raw;
  const int idx = blockIdx.x * blockDim.x + threadIdx.x;  // (diff_idx, k, lane)
  if (idx >= 256 * KF * 32) return;
  const int lane = idx & 31;
  const int k    = (idx >> 5) % KF;
  const int diff = (idx / (32 * KF)) << 4;
  const int h = lane >> 4, m = lane & 15;
  __bf16* dst = afrag + (size_t)idx * 16;
#pragma unroll
  for (int e = 0; e < 16; ++e) {
    const int kidx = (h ? 16 : 0) + e;
    const int lag = diff + m - kidx;
    float v = 0.0f;
    if (lag >= 0 && lag < SEQ_LEN && ((lag & 1) == 0))
      v = 2.0f * stu[(size_t)lag * KF + k];
    dst[e] = (__bf16)v;
  }
}

// ---------------------------------------------------------------------------
// Kernel 3: x_projT = (x @ M_inputs)^T  (bf16 WMMA, transposed bf16 store).
// ---------------------------------------------------------------------------
__global__ __launch_bounds__(256) void stu_proj_gemm(
    const float* __restrict__ x,                   // (B*L, D) fp32
    const unsigned short* __restrict__ mifrag_raw, // packed M_inputs
    unsigned short* __restrict__ xprojT_raw) {     // (B, D, L) bf16
  const __bf16* __restrict__ mifrag = (const __bf16*)mifrag_raw;
  __bf16* __restrict__ xprojT = (__bf16*)xprojT_raw;

  const int lane = threadIdx.x & 31;
  const int wave = threadIdx.x >> 5;
  const int tile = blockIdx.x * 8 + wave;
  const int tm = tile >> 6;                 // row tile over B*L (1024)
  const int tn = tile & 63;                 // col tile over D (64)
  const int h  = lane >> 4;
  const int m  = lane & 15;

  const float* xrow = x + (size_t)(tm * 16 + m) * DIM;
  const int dcol = tn * 16 + m;

  v8f acc = {};
  for (int j0 = 0; j0 < DIM; j0 += 32) {
    v16bf a;
    {
      const int kb = h ? 8 : 0;
      v4f f0 = *(const v4f*)(xrow + j0 + kb);
      v4f f1 = *(const v4f*)(xrow + j0 + kb + 4);
      v4f f2 = *(const v4f*)(xrow + j0 + 16 + kb);
      v4f f3 = *(const v4f*)(xrow + j0 + 16 + kb + 4);
#pragma unroll
      for (int e = 0; e < 4; ++e) {
        a[e]      = (__bf16)f0[e];
        a[4 + e]  = (__bf16)f1[e];
        a[8 + e]  = (__bf16)f2[e];
        a[12 + e] = (__bf16)f3[e];
      }
    }
    v16bf b = *(const v16bf*)(mifrag +
        ((size_t)tn * 32 + (j0 >> 5)) * 512 + (size_t)lane * 16);
    acc = __builtin_amdgcn_wmma_f32_16x16x32_bf16(false, a, false, b,
                                                  (short)0, acc, false, false);
  }

  const int rowA = tm * 16;
  const int b_   = rowA >> 12;
  const int t_   = (rowA & (SEQ_LEN - 1)) + (h ? 8 : 0);
  v8bf cv;
#pragma unroll
  for (int r = 0; r < 8; ++r) cv[r] = (__bf16)acc[r];
  *(v8bf*)(xprojT + ((size_t)b_ * DIM + dcol) * SEQ_LEN + t_) = cv;
}

// ---------------------------------------------------------------------------
// Kernel 4: fused rank-K Toeplitz convolution.
//   out[b,t,d] = sum_k M_filters[k,d] * (A_k @ x_proj[b])[t,d]
// Workgroup: (b, 16-row t-tile) x 256 channels; 8 waves x 2 d-tiles each, so
// every LDS A fragment feeds two WMMAs (ds:wmma = 1:1). A blocks are staged
// by TDM (or async b128) into a double-buffered 2x24 KB LDS region.
// ---------------------------------------------------------------------------
__global__ __launch_bounds__(256) void stu_conv(
    const unsigned short* __restrict__ xprojT_raw,  // (B, D, L) bf16
    const unsigned short* __restrict__ afrag_raw,   // A-fragment table
    const float* __restrict__ Mf,                   // (K, D) fp32
    float* __restrict__ out) {                      // (B, L, D) fp32
  const __bf16* __restrict__ xprojT = (const __bf16*)xprojT_raw;
  const __bf16* __restrict__ afrag  = (const __bf16*)afrag_raw;

  __shared__ __attribute__((aligned(128))) __bf16 Ablk[2][KF * 512];

  const int tid  = threadIdx.x;
  const int lane = tid & 31;
  const int wave = tid >> 5;
  const int bx   = blockIdx.x;
  const int dblk = bx & 3;                 // D/256 channel blocks
  const int tb   = bx >> 2;
  const int b    = tb >> 8;                // L/16 = 256 t-tiles
  const int t0   = (tb & 255) << 4;
  const int d0   = dblk * 256 + wave * 32;
  const int h = lane >> 4, m = lane & 15;
  const int dcolA = d0 + m;
  const int dcolB = d0 + 16 + m;

  // Per-lane channel scales for both d-tiles (registers; loops fully unrolled).
  float mfA[KF], mfB[KF];
#pragma unroll
  for (int k = 0; k < KF; ++k) {
    mfA[k] = Mf[(size_t)k * DIM + dcolA];
    mfB[k] = Mf[(size_t)k * DIM + dcolB];
  }

  const __bf16* xcolA = xprojT + ((size_t)b * DIM + dcolA) * SEQ_LEN;
  const __bf16* xcolB = xprojT + ((size_t)b * DIM + dcolB) * SEQ_LEN;

  int buf = 0;
  stage_ablk(afrag + ((size_t)(t0 >> 4)) * KF * 512, &Ablk[0][0], tid, wave);

  v8f accA = {}, accB = {};
  for (int s0 = 0; s0 <= t0; s0 += 32) {
    stage_wait();   // staged A block visible to all waves

    if (s0 + 32 <= t0)
      stage_ablk(afrag + ((size_t)((t0 - s0 - 32) >> 4)) * KF * 512,
                 &Ablk[buf ^ 1][0], tid, wave);

    // B fragments: 16 consecutive time steps -> one aligned 32B load each.
    const int so = s0 + (h ? 16 : 0);
    v16bf bfragA = *(const v16bf*)(xcolA + so);
    v16bf bfragB = *(const v16bf*)(xcolB + so);
    if (s0 + 32 <= t0) {
      __builtin_prefetch(xcolA + s0 + 32, 0, 1);
      __builtin_prefetch(xcolB + s0 + 32, 0, 1);
    }

#pragma unroll
    for (int k = 0; k < KF; ++k) {
      v16bf afr = *(const v16bf*)&Ablk[buf][k * 512 + lane * 16];  // ds_load
      v8f zA = {}, zB = {};
      zA = __builtin_amdgcn_wmma_f32_16x16x32_bf16(false, afr, false, bfragA,
                                                   (short)0, zA, false, false);
      zB = __builtin_amdgcn_wmma_f32_16x16x32_bf16(false, afr, false, bfragB,
                                                   (short)0, zB, false, false);
#pragma unroll
      for (int r = 0; r < 8; ++r) {
        accA[r] = fmaf(mfA[k], zA[r], accA[r]);
        accB[r] = fmaf(mfB[k], zB[r], accB[r]);
      }
    }
    buf ^= 1;
  }

  float* opA = out + ((size_t)b * SEQ_LEN + t0 + (h ? 8 : 0)) * DIM + dcolA;
  float* opB = out + ((size_t)b * SEQ_LEN + t0 + (h ? 8 : 0)) * DIM + dcolB;
#pragma unroll
  for (int r = 0; r < 8; ++r) {
    opA[(size_t)r * DIM] = accA[r];
    opB[(size_t)r * DIM] = accB[r];
  }
}

// ---------------------------------------------------------------------------
extern "C" void kernel_launch(void* const* d_in, const int* in_sizes, int n_in,
                              void* d_out, int out_size, void* d_ws, size_t ws_size,
                              hipStream_t stream) {
  if (ws_size < MIFRAG_OFF + MIFRAG_BYTES) return;  // ~42 MB scratch needed

  const float* x   = (const float*)d_in[0];  // (B, L, D)
  const float* stu = (const float*)d_in[1];  // (L, K)
  const float* Mi  = (const float*)d_in[2];  // (D, D)
  const float* Mf  = (const float*)d_in[3];  // (K, D)
  float* out = (float*)d_out;

  unsigned short* xprojT = (unsigned short*)d_ws;
  unsigned short* afrag  = (unsigned short*)((char*)d_ws + AFRAG_OFF);
  unsigned short* mifrag = (unsigned short*)((char*)d_ws + MIFRAG_OFF);

  stu_mifrag_build<<<(64 * 32 * 32) / 256, 256, 0, stream>>>(Mi, mifrag);
  stu_afrag_build<<<(256 * KF * 32) / 256, 256, 0, stream>>>(stu, afrag);
  stu_proj_gemm<<<(BATCH * SEQ_LEN / 16) * (DIM / 16) / 8, 256, 0, stream>>>(
      x, mifrag, xprojT);
  stu_conv<<<BATCH * (SEQ_LEN / 16) * (DIM / 256), 256, 0, stream>>>(
      xprojT, afrag, Mf, out);
}
